// NGCF_78494822302010
// MI455X (gfx1250) — matrix operand order repaired and checked by reading, not compiled, not run on hardware
//
#include <hip/hip_runtime.h>
#include <hip/hip_bf16.h>

typedef __attribute__((ext_vector_type(16))) __bf16 v16bf;
typedef __attribute__((ext_vector_type(8)))  float  v8f;

#define NB      8      // batches
#define NNODES  4096   // nodes per batch (2*2048)
#define NPER    2048
#define DFEAT   64

// ---------------------------------------------------------------------------
// 1) Gather: ego[b][n][d] (f32) and egoT[b][d][n] (bf16, K-contiguous for WMMA B)
// ---------------------------------------------------------------------------
__global__ __launch_bounds__(256) void ngcf_gather(
    const int* __restrict__ uidx, const int* __restrict__ iidx,
    const float* __restrict__ utab, const float* __restrict__ itab,
    float* __restrict__ ego, __bf16* __restrict__ egoT)
{
    int t = blockIdx.x * 256 + threadIdx.x;       // over B*N*D = 2M
    int e        = t & 63;
    int node_lin = t >> 6;                        // b*4096 + n
    int n        = node_lin & (NNODES - 1);
    int b        = node_lin >> 12;
    int idx;
    const float* tab;
    if (n < NPER) { idx = uidx[b * NPER + n];          tab = utab; }
    else          { idx = iidx[b * NPER + (n - NPER)]; tab = itab; }
    float v = tab[(size_t)idx * DFEAT + e];
    ego[(size_t)t] = v;
    egoT[(((size_t)b * DFEAT) + e) * NNODES + n] = (__bf16)v;
}

// ---------------------------------------------------------------------------
// 2) side = adj @ ego via v_wmma_f32_16x16x32_bf16, register double-buffered.
//    grid.x = B * (N/128); 8 waves/WG, wave -> 16 rows x 64 cols tile.
// ---------------------------------------------------------------------------
__global__ __launch_bounds__(256) void ngcf_gemm_side(
    const float* __restrict__ adj, const __bf16* __restrict__ egoT,
    float* __restrict__ side)
{
    const int b      = blockIdx.x >> 5;           // 32 row-blocks per batch
    const int rowblk = blockIdx.x & 31;
    const int wave   = threadIdx.x >> 5;
    const int lane   = threadIdx.x & 31;
    const int row0   = rowblk * 128 + wave * 16;
    const int m      = lane & 15;
    const int hi     = lane >> 4;                 // 0 or 1 (half-wave)

    const float*  arow = adj + ((size_t)b * NNODES + (row0 + m)) * NNODES;
    const __bf16* egb  = egoT + (size_t)b * DFEAT * NNODES;
    // B-fragment base pointers for the four 16-col tiles (contiguous K per lane)
    const __bf16* bp0 = egb + (size_t)( 0 + m) * NNODES + hi * 16;
    const __bf16* bp1 = egb + (size_t)(16 + m) * NNODES + hi * 16;
    const __bf16* bp2 = egb + (size_t)(32 + m) * NNODES + hi * 16;
    const __bf16* bp3 = egb + (size_t)(48 + m) * NNODES + hi * 16;
    const int abase = hi * 16;                    // A layout: hi half owns K+16 .. K+31

    v8f acc0 = {}, acc1 = {}, acc2 = {}, acc3 = {};

    // ---- software pipeline registers (next k-chunk)
    v8f   a0n, a1n;
    v16bf b0n, b1n, b2n, b3n;

    auto load_chunk = [&](int k) {
        a0n = *(const v8f*)(arow + k + abase);        // K = k+base .. +7
        a1n = *(const v8f*)(arow + k + abase + 8);    // K = k+base+8 .. +15
        b0n = *(const v16bf*)(bp0 + k);
        b1n = *(const v16bf*)(bp1 + k);
        b2n = *(const v16bf*)(bp2 + k);
        b3n = *(const v16bf*)(bp3 + k);
    };
    auto compute = [&](const v8f& a0c, const v8f& a1c,
                       const v16bf& B0, const v16bf& B1,
                       const v16bf& B2, const v16bf& B3) {
        v16bf a;
        #pragma unroll
        for (int i = 0; i < 8; ++i) {
            a[i]     = (__bf16)a0c[i];
            a[8 + i] = (__bf16)a1c[i];
        }
        acc0 = __builtin_amdgcn_wmma_f32_16x16x32_bf16(false, a, false, B0, (short)0, acc0, false, false);
        acc1 = __builtin_amdgcn_wmma_f32_16x16x32_bf16(false, a, false, B1, (short)0, acc1, false, false);
        acc2 = __builtin_amdgcn_wmma_f32_16x16x32_bf16(false, a, false, B2, (short)0, acc2, false, false);
        acc3 = __builtin_amdgcn_wmma_f32_16x16x32_bf16(false, a, false, B3, (short)0, acc3, false, false);
    };

    load_chunk(0);                                    // prologue

    for (int k = 0; k < NNODES - 32; k += 32) {
        // rotate next -> current
        v8f   a0c = a0n, a1c = a1n;
        v16bf B0 = b0n, B1 = b1n, B2 = b2n, B3 = b3n;
        // issue next chunk's loads a full iteration ahead of their use
        load_chunk(k + 32);
        // keep the HBM adj stream ahead of the XDL pipe (-> global_prefetch_b8)
        __builtin_prefetch(arow + k + abase + 1024, 0, 3);
        __builtin_prefetch(arow + k + abase + 1040, 0, 3);
        compute(a0c, a1c, B0, B1, B2, B3);
    }
    compute(a0n, a1n, b0n, b1n, b2n, b3n);            // epilogue

    // ---- store: C layout -> VGPR r holds row (hi*8+r), col = m within tile
    float* out = side + ((size_t)b * NNODES + row0) * DFEAT;
    #pragma unroll
    for (int r = 0; r < 8; ++r) {
        size_t ro = (size_t)(hi * 8 + r) * DFEAT;
        out[ro +  0 + m] = acc0[r];
        out[ro + 16 + m] = acc1[r];
        out[ro + 32 + m] = acc2[r];
        out[ro + 48 + m] = acc3[r];
    }
}

// ---------------------------------------------------------------------------
// 3) ego = lrelu(side@gcWt + gcb) + lrelu((ego*side)@biWt + bib); refresh egoT.
//    Weights staged transposed in LDS (conflict-free inner loop).
// ---------------------------------------------------------------------------
__global__ __launch_bounds__(256) void ngcf_pointwise(
    const float* __restrict__ side, float* __restrict__ ego,
    __bf16* __restrict__ egoT,
    const float* __restrict__ gcw, const float* __restrict__ gcb,
    const float* __restrict__ biw, const float* __restrict__ bib)
{
    __shared__ float s_gcw[64 * 64];   // [d][e] = gcw[e][d]
    __shared__ float s_biw[64 * 64];
    __shared__ float s_side[4][64];
    __shared__ float s_prod[4][64];
    __shared__ float s_gcb[64], s_bib[64];

    const int tid = threadIdx.x;
    for (int i = tid; i < 4096; i += 256) {
        int e = i >> 6, d = i & 63;
        s_gcw[d * 64 + e] = gcw[i];
        s_biw[d * 64 + e] = biw[i];
    }
    if (tid < 64) { s_gcb[tid] = gcb[tid]; s_bib[tid] = bib[tid]; }

    const int e = tid & 63;
    const int r = tid >> 6;                        // 4 rows per block
    const size_t row = (size_t)blockIdx.x * 4 + r; // over B*N rows
    const float sv = side[row * 64 + e];
    const float ev = ego [row * 64 + e];
    s_side[r][e] = sv;
    s_prod[r][e] = sv * ev;
    __syncthreads();

    float acc1 = s_gcb[e];
    float acc2 = s_bib[e];
    #pragma unroll
    for (int d = 0; d < 64; ++d) {
        acc1 += s_side[r][d] * s_gcw[d * 64 + e];
        acc2 += s_prod[r][d] * s_biw[d * 64 + e];
    }
    float o1 = (acc1 >= 0.0f) ? acc1 : 0.01f * acc1;   // leaky_relu
    float o2 = (acc2 >= 0.0f) ? acc2 : 0.01f * acc2;
    float o  = o1 + o2;

    ego[row * 64 + e] = o;
    size_t b_ = row >> 12;
    size_t n  = row & (NNODES - 1);
    egoT[((b_ << 6) + e) * NNODES + n] = (__bf16)o;
}

// ---------------------------------------------------------------------------
// 4) scores + log_softmax over singleton axis (== s - s)
// ---------------------------------------------------------------------------
__global__ __launch_bounds__(256) void ngcf_final(
    const float* __restrict__ ego, float* __restrict__ out)
{
    int t = blockIdx.x * 256 + threadIdx.x;      // 16384 pairs
    int b = t >> 11;
    int j = t & (NPER - 1);
    const float* u = ego + ((size_t)b * NNODES + j) * DFEAT;
    const float* v = ego + ((size_t)b * NNODES + NPER + j) * DFEAT;
    float s = 0.0f;
    #pragma unroll
    for (int d = 0; d < DFEAT; ++d) s += u[d] * v[d];
    out[t] = s - s;   // log_softmax over a size-1 axis: x - logsumexp(x) = x - x
}

// ---------------------------------------------------------------------------
extern "C" void kernel_launch(void* const* d_in, const int* in_sizes, int n_in,
                              void* d_out, int out_size, void* d_ws, size_t ws_size,
                              hipStream_t stream)
{
    const int*   uidx = (const int*)  d_in[0];
    const int*   iidx = (const int*)  d_in[1];
    const float* adj  = (const float*)d_in[2];
    const float* utab = (const float*)d_in[3];
    const float* itab = (const float*)d_in[4];
    const float* gcw  = (const float*)d_in[5];
    const float* gcb  = (const float*)d_in[6];
    const float* biw  = (const float*)d_in[7];
    const float* bib  = (const float*)d_in[8];
    float* out = (float*)d_out;

    char* ws = (char*)d_ws;
    float*  ego  = (float*) (ws);                        // 8 MB
    float*  side = (float*) (ws + (size_t)(8  << 20));   // 8 MB
    __bf16* egoT = (__bf16*)(ws + (size_t)(16 << 20));   // 4 MB

    ngcf_gather<<<8192, 256, 0, stream>>>(uidx, iidx, utab, itab, ego, egoT);

    for (int l = 0; l < 3; ++l) {
        ngcf_gemm_side<<<NB * (NNODES / 128), 256, 0, stream>>>(adj, egoT, side);
        ngcf_pointwise<<<(NB * NNODES) / 4, 256, 0, stream>>>(
            side, ego, egoT,
            gcw + (size_t)l * 64 * 64, gcb + (size_t)l * 64,
            biw + (size_t)l * 64 * 64, bib + (size_t)l * 64);
    }

    ngcf_final<<<(NB * NPER) / 256, 256, 0, stream>>>(ego, out);
}
